// TemporalCfCEncoder_4398046511330
// MI455X (gfx1250) — matrix operand order repaired and checked by reading, not compiled, not run on hardware
//
#include <hip/hip_runtime.h>

// ---------------------------------------------------------------------------
// Problem constants (reference: T=128, N=256, D=256, L=3, H=4)
// ---------------------------------------------------------------------------
constexpr int cT = 128;
constexpr int cN = 256;
constexpr int cD = 256;
constexpr int cH = 4;
constexpr int cL = 3;
constexpr int cHD = cD / cH;           // 64
constexpr size_t cR = (size_t)cT * cN; // 32768 rows

typedef __bf16 bf16_t;
typedef __attribute__((ext_vector_type(16))) __bf16 v16bf;
typedef __attribute__((ext_vector_type(8))) float v8f;
typedef __attribute__((ext_vector_type(4))) unsigned int u32x4;
typedef __attribute__((ext_vector_type(8))) int i32x8;
typedef __attribute__((ext_vector_type(4))) int i32x4;

#if defined(__has_builtin)
#if __has_builtin(__builtin_amdgcn_tensor_load_to_lds)
#define HAVE_TDM 1
#else
#define HAVE_TDM 0
#endif
#else
#define HAVE_TDM 0
#endif

// ---------------------------------------------------------------------------
// Tensor Data Mover: 2D tile (rows x cols, bf16) global -> LDS.
// D# built per CDNA5 ISA 8.3/8.4: group0 = {flags/count, lds_addr, gaddr,
// gaddr_hi|type=2}; group1 packs data_size=2B, tensor_dim0/1, tile_dim0/1,
// dim0 stride. Groups 2/3 zero (2D tensor). Issued by one wave; EXEC ignored.
// ---------------------------------------------------------------------------
__device__ __forceinline__ void tdm_load_2d_bf16(unsigned lds_off_bytes,
                                                 const bf16_t* gsrc,
                                                 unsigned rows, unsigned cols,
                                                 unsigned stride_elems) {
#if HAVE_TDM
  unsigned long long ga = (unsigned long long)(uintptr_t)gsrc;
  u32x4 g0;
  g0[0] = 1u;                                   // count=1, user descriptor
  g0[1] = lds_off_bytes;                        // lds_addr
  g0[2] = (unsigned)(ga & 0xFFFFFFFFu);         // global_addr[31:0]
  g0[3] = (unsigned)((ga >> 32) & 0x01FFFFFFu)  // global_addr[56:32]
          | (2u << 30);                         // type = 2 ("image")
  i32x8 g1;
  g1[0] = (int)(1u << 16);                      // data_size=1 -> 2 bytes
  g1[1] = (int)((cols & 0xFFFFu) << 16);        // tensor_dim0[15:0]
  g1[2] = (int)(((cols >> 16) & 0xFFFFu) | ((rows & 0xFFFFu) << 16));
  g1[3] = (int)(((rows >> 16) & 0xFFFFu) | ((cols & 0xFFFFu) << 16)); // tile_dim0
  g1[4] = (int)(rows & 0xFFFFu);                // tile_dim1 (tile_dim2=0)
  g1[5] = (int)stride_elems;                    // tensor_dim0_stride[31:0]
  g1[6] = 0;
  g1[7] = 0;
  i32x4 z4 = {0, 0, 0, 0};
#if __clang_major__ >= 23
  i32x8 z8 = {0, 0, 0, 0, 0, 0, 0, 0};
  __builtin_amdgcn_tensor_load_to_lds(g0, g1, z4, z4, z8, 0);
#else
  __builtin_amdgcn_tensor_load_to_lds(g0, g1, z4, z4, 0);
#endif
#else
  (void)lds_off_bytes; (void)gsrc; (void)rows; (void)cols; (void)stride_elems;
#endif
}

__device__ __forceinline__ void wait_tensorcnt0() {
#if defined(__has_builtin) && __has_builtin(__builtin_amdgcn_s_wait_tensorcnt)
  __builtin_amdgcn_s_wait_tensorcnt(0);
#else
  asm volatile("s_wait_tensorcnt 0x0" ::: "memory");
#endif
}

// ---------------------------------------------------------------------------
// WMMA helper: D = A(16x32 bf16) * B(32x16 bf16) + C(16x16 f32)
// ---------------------------------------------------------------------------
__device__ __forceinline__ v8f wmma_bf16(v16bf a, v16bf b, v8f c) {
  return __builtin_amdgcn_wmma_f32_16x16x32_bf16(
      /*neg_a=*/false, a, /*neg_b=*/false, b,
      /*c_mod=*/(short)0, c, /*reuse_a=*/false, /*reuse_b=*/false);
}

// A-matrix tile (16x32, bf16) from row-major [row][k] storage.
// Layout: lanes 0-15 row m=lane, K = k0..k0+15;
//         lanes 16-31 row m=lane-16, K = k0+16..k0+31 (contiguous, matches B).
__device__ __forceinline__ v16bf load_tileA(const bf16_t* __restrict__ base,
                                            int ld, int row0, int k0) {
  int lane = threadIdx.x & 31;
  int r = row0 + (lane & 15);
  int kk = k0 + ((lane >> 4) << 4);
  const bf16_t* p = base + (size_t)r * ld + kk;
  union { uint4 u[2]; v16bf v; } cv;
  cv.u[0] = *(const uint4*)(p);
  cv.u[1] = *(const uint4*)(p + 8);
  return cv.v;
}

// B-matrix tile (32x16, bf16) where B[k][n] = W[n][k], W row-major [out][in]
// (computing X @ W^T). Lane holds column n=lane%16; half 0 K=k0..k0+15,
// half 1 K=k0+16..k0+31 (contiguous, same order as load_tileA).
__device__ __forceinline__ v16bf load_tileB(const bf16_t* __restrict__ base,
                                            int ld, int row0, int k0) {
  int lane = threadIdx.x & 31;
  int r = row0 + (lane & 15);
  int kk = k0 + ((lane >> 4) << 4);
  const bf16_t* p = base + (size_t)r * ld + kk;
  union { uint4 u[2]; v16bf v; } cv;
  cv.u[0] = *(const uint4*)(p);
  cv.u[1] = *(const uint4*)(p + 8);
  return cv.v;
}

// B-matrix tile where B[k][n] is stored k-major with stride ldk (gather).
__device__ __forceinline__ v16bf load_tileB_strided(
    const bf16_t* __restrict__ base, int ldk, int n0, int k0) {
  int lane = threadIdx.x & 31;
  int n = n0 + (lane & 15);
  int kk = k0 + ((lane >> 4) << 4);
  v16bf v;
#pragma unroll
  for (int e = 0; e < 16; ++e) v[e] = base[(size_t)(kk + e) * ldk + n];
  return v;
}

// ---------------------------------------------------------------------------
// Generic bf16 WMMA GEMM with fused epilogue.
//   out[M,Nout] = act( A[M,K] @ W[Nout,K]^T + bias ) (+ resid)
// wave tile = 16(M) x 64(N). ACT: 0=none, 1=silu. OUT_BF16 selects out dtype.
// ---------------------------------------------------------------------------
template <int ACT, bool OUT_BF16, bool RESID>
__global__ __launch_bounds__(256) void gemm_bf16_kernel(
    const bf16_t* __restrict__ A, int lda,
    const bf16_t* __restrict__ W, int ldw,
    const float* __restrict__ bias,
    const float* __restrict__ resid, int ldr,
    void* __restrict__ outp, int ldo,
    int M, int Nout, int K) {
  int wave = blockIdx.x * (blockDim.x >> 5) + (threadIdx.x >> 5);
  int tiles_n = Nout >> 6;
  int tm = wave / tiles_n;
  int tn = wave % tiles_n;
  int row0 = tm << 4;
  if (row0 >= M) return;
  int lane = threadIdx.x & 31;

  v8f acc[4] = {};
  for (int k0 = 0; k0 < K; k0 += 32) {
    v16bf a = load_tileA(A, lda, row0, k0);
    if (k0 + 32 < K)
      __builtin_prefetch(A + (size_t)(row0 + (lane & 15)) * lda + k0 + 32, 0, 1);
#pragma unroll
    for (int j = 0; j < 4; ++j) {
      v16bf b = load_tileB(W, ldw, (tn << 6) + (j << 4), k0);
      acc[j] = wmma_bf16(a, b, acc[j]);
    }
  }

  int mh = (lane >> 4) << 3;
#pragma unroll
  for (int j = 0; j < 4; ++j) {
    int n = (tn << 6) + (j << 4) + (lane & 15);
    float bv = bias ? bias[n] : 0.0f;
#pragma unroll
    for (int c = 0; c < 8; ++c) {
      int row = row0 + mh + c;
      float v = acc[j][c] + bv;
      if (ACT == 1) v = v * (1.0f / (1.0f + __expf(-v)));  // silu
      if (RESID) v += resid[(size_t)row * ldr + n];
      if (OUT_BF16)
        ((bf16_t*)outp)[(size_t)row * ldo + n] = (bf16_t)v;
      else
        ((float*)outp)[(size_t)row * ldo + n] = v;
    }
  }
}

// ---------------------------------------------------------------------------
// Multi-head attention, one workgroup per (t, h). Flash-style online softmax.
// qkv: [T*N, 768] bf16 (q | k | v per row). out: [T*N, 256] bf16.
// V head-slice (256x64) is DMA'd to LDS by the TDM once per block, so the
// inner-loop B gather for P@V is ds traffic instead of strided global.
// ---------------------------------------------------------------------------
__global__ __launch_bounds__(256) void attn_kernel(
    const bf16_t* __restrict__ qkv, bf16_t* __restrict__ out) {
  // [0, 16384): V slice (256 x 64);  [16384, 16384+8*512): P staging per wave
  __shared__ __align__(16) bf16_t smem[16384 + 8 * 512];
  int t = blockIdx.x >> 2;
  int h = blockIdx.x & 3;
  int wave = threadIdx.x >> 5;
  int lane = threadIdx.x & 31;
  const bf16_t* qb = qkv + (size_t)t * cN * (3 * cD) + h * cHD;
  const bf16_t* kbp = qb + cD;
  const bf16_t* vbp = qb + 2 * cD;
  bf16_t* vs = smem;
  bf16_t* pw = smem + 16384 + wave * 512;

#if HAVE_TDM
  if (wave == 0) {
    tdm_load_2d_bf16(/*lds_off=*/0, vbp, /*rows=*/cN, /*cols=*/cHD,
                     /*stride=*/3 * cD);
    wait_tensorcnt0();
  }
#else
  {  // cooperative copy fallback: 256 rows x 64 elems = 2048 16B chunks
    for (int i = threadIdx.x; i < 2048; i += 256) {
      int row = i >> 3, cc = (i & 7) * 8;
      *(uint4*)(vs + row * cHD + cc) =
          *(const uint4*)(vbp + (size_t)row * 3 * cD + cc);
    }
  }
#endif
  __syncthreads();

  v8f o[2][4] = {};
  float mi[2][8], li[2][8];
#pragma unroll
  for (int q = 0; q < 2; ++q)
#pragma unroll
    for (int j = 0; j < 8; ++j) { mi[q][j] = -1e30f; li[q][j] = 0.0f; }

  for (int kb = 0; kb < cN; kb += 32) {
#pragma unroll
    for (int q = 0; q < 2; ++q) {
      int q0 = wave * 32 + q * 16;
      v8f s0 = {}, s1 = {};
#pragma unroll
      for (int k0 = 0; k0 < cHD; k0 += 32) {
        v16bf a = load_tileA(qb, 3 * cD, q0, k0);
        s0 = wmma_bf16(a, load_tileB(kbp, 3 * cD, kb, k0), s0);
        s1 = wmma_bf16(a, load_tileB(kbp, 3 * cD, kb + 16, k0), s1);
      }
      const float sc = 0.125f;  // 1/sqrt(64)
      float al[8];
#pragma unroll
      for (int j = 0; j < 8; ++j) {
        float v0 = s0[j] * sc, v1 = s1[j] * sc;
        float cm = fmaxf(v0, v1);
#pragma unroll
        for (int off = 1; off < 16; off <<= 1)
          cm = fmaxf(cm, __shfl_xor(cm, off, 32));
        float nm = fmaxf(mi[q][j], cm);
        al[j] = __expf(mi[q][j] - nm);
        float p0 = __expf(v0 - nm);
        float p1 = __expf(v1 - nm);
        s0[j] = p0; s1[j] = p1;
        float ps = p0 + p1;
#pragma unroll
        for (int off = 1; off < 16; off <<= 1)
          ps += __shfl_xor(ps, off, 32);
        li[q][j] = li[q][j] * al[j] + ps;
        mi[q][j] = nm;
      }
#pragma unroll
      for (int jn = 0; jn < 4; ++jn)
#pragma unroll
        for (int j = 0; j < 8; ++j) o[q][jn][j] *= al[j];

      // Stage P (16x32) to LDS in row-major, reload in A-layout.
      int mh = (lane >> 4) << 3;
      int nn = lane & 15;
#pragma unroll
      for (int j = 0; j < 8; ++j) {
        pw[(mh + j) * 32 + nn] = (bf16_t)s0[j];
        pw[(mh + j) * 32 + 16 + nn] = (bf16_t)s1[j];
      }
      __syncthreads();
      v16bf pa = load_tileA(pw, 32, 0, 0);
#pragma unroll
      for (int jn = 0; jn < 4; ++jn) {
        v16bf vb = load_tileB_strided(vs, cHD, jn * 16, kb);
        o[q][jn] = wmma_bf16(pa, vb, o[q][jn]);
      }
      __syncthreads();
    }
  }

  int mh = (lane >> 4) << 3;
#pragma unroll
  for (int q = 0; q < 2; ++q) {
    int q0 = wave * 32 + q * 16;
#pragma unroll
    for (int jn = 0; jn < 4; ++jn) {
      int n = h * cHD + jn * 16 + (lane & 15);
#pragma unroll
      for (int j = 0; j < 8; ++j) {
        float v = o[q][jn][j] / li[q][j];
        out[(size_t)(t * cN + q0 + mh + j) * cD + n] = (bf16_t)v;
      }
    }
  }
}

// ---------------------------------------------------------------------------
// LayerNorm over last dim (256) -> bf16. One row per workgroup, LDS tree.
// ---------------------------------------------------------------------------
__global__ __launch_bounds__(256) void ln_kernel(
    const float* __restrict__ x, const float* __restrict__ s,
    const float* __restrict__ b, bf16_t* __restrict__ out) {
  __shared__ float red[256];
  int r = blockIdx.x, c = threadIdx.x;
  float v = x[(size_t)r * cD + c];
  red[c] = v;
  __syncthreads();
  for (int off = 128; off > 0; off >>= 1) {
    if (c < off) red[c] += red[c + off];
    __syncthreads();
  }
  float mean = red[0] * (1.0f / cD);
  __syncthreads();
  float d = v - mean;
  red[c] = d * d;
  __syncthreads();
  for (int off = 128; off > 0; off >>= 1) {
    if (c < off) red[c] += red[c + off];
    __syncthreads();
  }
  float var = red[0] * (1.0f / cD);
  float y = d * rsqrtf(var + 1e-5f) * s[c] + b[c];
  out[(size_t)r * cD + c] = (bf16_t)y;
}

// ---------------------------------------------------------------------------
// One CfC recurrence step. xh = [x_t | h] handled as two K-halves (no concat).
// The 32-row x_t and h slabs for this block are DMA'd to LDS via the TDM, so
// the K-loop A-side traffic is ds_load_b128 (weights stream from L2).
// Fuses the three [256,512]x[512,256] GEMMs + tanh/tanh/sigmoid gate.
// ---------------------------------------------------------------------------
__global__ __launch_bounds__(256) void cell_step_kernel(
    const bf16_t* __restrict__ xs,     // [T*N, D] bf16
    const bf16_t* __restrict__ hprev,  // [N, D] bf16
    const bf16_t* __restrict__ w1, const float* __restrict__ b1,
    const bf16_t* __restrict__ w2, const float* __restrict__ b2,
    const bf16_t* __restrict__ wt, const float* __restrict__ bt,
    const float* __restrict__ logtau, const float* __restrict__ dts, int t,
    const float* __restrict__ seq_in,  // nullable (layer 0)
    float* __restrict__ seq_out, bf16_t* __restrict__ hnext) {
  // [0, 32*256): x_t slab; [32*256, 64*256): h slab
  __shared__ __align__(16) bf16_t asmem[64 * cD];
  int widx = threadIdx.x >> 5;
  int wave = blockIdx.x * 8 + widx;
  int tn = wave & 3;            // 4 tiles of 64 in N(out)
  int tml = (wave >> 2) & 1;    // local M tile within block (0/1)
  int row0_blk = blockIdx.x * 32;
  int row0 = row0_blk + tml * 16;
  int lane = threadIdx.x & 31;
  float dt = dts[t];
  const bf16_t* xbase = xs + (size_t)t * cN * cD;

#if HAVE_TDM
  if (widx == 0) {
    tdm_load_2d_bf16(0, xbase + (size_t)row0_blk * cD, 32, cD, cD);
    tdm_load_2d_bf16(32 * cD * 2, hprev + (size_t)row0_blk * cD, 32, cD, cD);
    wait_tensorcnt0();
  }
#else
  {  // cooperative copy fallback: 2 slabs x 32 rows x 256 elems
    const uint4* gx = (const uint4*)(xbase + (size_t)row0_blk * cD);
    const uint4* gh = (const uint4*)(hprev + (size_t)row0_blk * cD);
    uint4* sx = (uint4*)asmem;
    uint4* sh = (uint4*)(asmem + 32 * cD);
    for (int i = threadIdx.x; i < 1024; i += 256) { sx[i] = gx[i]; sh[i] = gh[i]; }
  }
#endif
  __syncthreads();

  v8f a1[4] = {}, a2[4] = {}, ag[4] = {};
  for (int k0 = 0; k0 < 2 * cD; k0 += 32) {
    v16bf a = (k0 < cD) ? load_tileA(asmem, cD, tml * 16, k0)
                        : load_tileA(asmem + 32 * cD, cD, tml * 16, k0 - cD);
#pragma unroll
    for (int j = 0; j < 4; ++j) {
      int nb = (tn << 6) + (j << 4);
      a1[j] = wmma_bf16(a, load_tileB(w1, 2 * cD, nb, k0), a1[j]);
      a2[j] = wmma_bf16(a, load_tileB(w2, 2 * cD, nb, k0), a2[j]);
      ag[j] = wmma_bf16(a, load_tileB(wt, 2 * cD, nb, k0), ag[j]);
    }
  }

  int mh = (lane >> 4) << 3;
#pragma unroll
  for (int j = 0; j < 4; ++j) {
    int n = (tn << 6) + (j << 4) + (lane & 15);
    float tau = __expf(logtau[n]);
    float b1v = b1[n], b2v = b2[n], btv = bt[n];
#pragma unroll
    for (int c = 0; c < 8; ++c) {
      int row = row0 + mh + c;
      float f1 = tanhf(a1[j][c] + b1v);
      float f2 = tanhf(a2[j][c] + b2v);
      // sigmoid(-tau*dt + z) = 1 / (1 + exp(tau*dt - z))
      float g = 1.0f / (1.0f + __expf(tau * dt - (ag[j][c] + btv)));
      float hv = g * f1 + (1.0f - g) * f2;
      size_t oi = (size_t)t * cN * cD + (size_t)row * cD + n;
      seq_out[oi] = hv + (seq_in ? seq_in[oi] : 0.0f);
      hnext[(size_t)row * cD + n] = (bf16_t)hv;
    }
  }
}

// ---------------------------------------------------------------------------
// Small utility kernels
// ---------------------------------------------------------------------------
__global__ void prep_kernel(const float* __restrict__ sensor_time,
                            const float* __restrict__ re_proj_w,
                            const float* __restrict__ re_proj_b,
                            const float* __restrict__ re_norm,
                            const float* __restrict__ tb_ff2_b,
                            float* __restrict__ dts,
                            float* __restrict__ ff2b) {
  int i = threadIdx.x;
  if (i < cT)
    dts[i] = (i == 0) ? sensor_time[0] : sensor_time[i] - sensor_time[i - 1];
  if (i < cD) {
    float rb = re_proj_w[i] * re_norm[0] + re_proj_b[i];
    ff2b[i] = tb_ff2_b[i] + rb;  // fold re_bias into FF2 bias
  }
}

__global__ void cvt_kernel(const float* __restrict__ src,
                           bf16_t* __restrict__ dst, int n) {
  int i = blockIdx.x * blockDim.x + threadIdx.x;
  if (i < n) dst[i] = (bf16_t)src[i];
}

__global__ void zero_bf16_kernel(bf16_t* __restrict__ p, int n) {
  int i = blockIdx.x * blockDim.x + threadIdx.x;
  if (i < n) p[i] = (bf16_t)0.0f;
}

// ---------------------------------------------------------------------------
// Host launcher
// ---------------------------------------------------------------------------
extern "C" void kernel_launch(void* const* d_in, const int* in_sizes, int n_in,
                              void* d_out, int out_size, void* d_ws,
                              size_t ws_size, hipStream_t stream) {
  (void)in_sizes; (void)n_in; (void)out_size; (void)ws_size;
  const float* spatial     = (const float*)d_in[0];
  const float* re_norm     = (const float*)d_in[1];
  const float* sensor_time = (const float*)d_in[2];
  const float* re_proj_w   = (const float*)d_in[3];
  const float* re_proj_b   = (const float*)d_in[4];
  const float* tb_n1_s     = (const float*)d_in[5];
  const float* tb_n1_b     = (const float*)d_in[6];
  const float* tb_in_w     = (const float*)d_in[7];
  const float* tb_in_b     = (const float*)d_in[8];
  const float* tb_out_w    = (const float*)d_in[9];
  const float* tb_out_b    = (const float*)d_in[10];
  const float* tb_n2_s     = (const float*)d_in[11];
  const float* tb_n2_b     = (const float*)d_in[12];
  const float* tb_ff1_w    = (const float*)d_in[13];
  const float* tb_ff1_b    = (const float*)d_in[14];
  const float* tb_ff2_w    = (const float*)d_in[15];
  const float* tb_ff2_b    = (const float*)d_in[16];
  const float* cell_ff1_w  = (const float*)d_in[17];
  const float* cell_ff1_b  = (const float*)d_in[18];
  const float* cell_ff2_w  = (const float*)d_in[19];
  const float* cell_ff2_b  = (const float*)d_in[20];
  const float* cell_logtau = (const float*)d_in[21];
  const float* cell_tb_w   = (const float*)d_in[22];
  const float* cell_tb_b   = (const float*)d_in[23];

  // Carve workspace
  size_t off = 0;
  auto carve = [&](size_t bytes) -> void* {
    off = (off + 255) & ~(size_t)255;
    void* p = (char*)d_ws + off;
    off += bytes;
    return p;
  };
  float*  dts   = (float*)carve(cT * 4);
  float*  ff2b  = (float*)carve(cD * 4);
  bf16_t* w_in  = (bf16_t*)carve((size_t)3 * cD * cD * 2);
  bf16_t* w_out = (bf16_t*)carve((size_t)cD * cD * 2);
  bf16_t* w_f1  = (bf16_t*)carve((size_t)2 * cD * cD * 2);
  bf16_t* w_f2  = (bf16_t*)carve((size_t)cD * 2 * cD * 2);
  bf16_t* w_c1  = (bf16_t*)carve((size_t)cL * cD * 2 * cD * 2);
  bf16_t* w_c2  = (bf16_t*)carve((size_t)cL * cD * 2 * cD * 2);
  bf16_t* w_ct  = (bf16_t*)carve((size_t)cL * cD * 2 * cD * 2);
  bf16_t* xln   = (bf16_t*)carve(cR * cD * 2);
  bf16_t* qkvb  = (bf16_t*)carve(cR * 3 * cD * 2);
  bf16_t* attno = (bf16_t*)carve(cR * cD * 2);
  float*  xr    = (float*)carve(cR * cD * 4);
  bf16_t* y2    = (bf16_t*)carve(cR * cD * 2);
  bf16_t* ffh   = (bf16_t*)carve(cR * 2 * cD * 2);
  bf16_t* xsb   = (bf16_t*)carve(cR * cD * 2);
  float*  seqA  = (float*)carve(cR * cD * 4);
  float*  seqB  = (float*)carve(cR * cD * 4);
  bf16_t* hA    = (bf16_t*)carve((size_t)cN * cD * 2);
  bf16_t* hB    = (bf16_t*)carve((size_t)cN * cD * 2);

  prep_kernel<<<1, 256, 0, stream>>>(sensor_time, re_proj_w, re_proj_b,
                                     re_norm, tb_ff2_b, dts, ff2b);
  auto cvt = [&](const float* s, bf16_t* d, int n) {
    cvt_kernel<<<(n + 255) / 256, 256, 0, stream>>>(s, d, n);
  };
  cvt(tb_in_w, w_in, 3 * cD * cD);
  cvt(tb_out_w, w_out, cD * cD);
  cvt(tb_ff1_w, w_f1, 2 * cD * cD);
  cvt(tb_ff2_w, w_f2, cD * 2 * cD);
  cvt(cell_ff1_w, w_c1, cL * cD * 2 * cD);
  cvt(cell_ff2_w, w_c2, cL * cD * 2 * cD);
  cvt(cell_tb_w, w_ct, cL * cD * 2 * cD);

  const int mt = (int)(cR / 16);  // 2048 M tiles
  for (int l = 0; l < cL; ++l) {
    const float* sin = (l == 0) ? spatial : ((l == 1) ? seqA : seqB);
    float* sout = (l == 0) ? seqA : ((l == 1) ? seqB : (float*)d_out);

    // ---- token block (parallel over T) ----
    ln_kernel<<<(unsigned)cR, 256, 0, stream>>>(sin, tb_n1_s, tb_n1_b, xln);
    gemm_bf16_kernel<0, true, false>
        <<<mt * (3 * cD / 64) / 8, 256, 0, stream>>>(
            xln, cD, w_in, cD, tb_in_b, nullptr, 0, qkvb, 3 * cD,
            (int)cR, 3 * cD, cD);
    attn_kernel<<<cT * cH, 256, 0, stream>>>(qkvb, attno);
    gemm_bf16_kernel<0, false, true>
        <<<mt * (cD / 64) / 8, 256, 0, stream>>>(
            attno, cD, w_out, cD, tb_out_b, sin, cD, xr, cD,
            (int)cR, cD, cD);
    ln_kernel<<<(unsigned)cR, 256, 0, stream>>>(xr, tb_n2_s, tb_n2_b, y2);
    gemm_bf16_kernel<1, true, false>
        <<<mt * (2 * cD / 64) / 8, 256, 0, stream>>>(
            y2, cD, w_f1, cD, tb_ff1_b, nullptr, 0, ffh, 2 * cD,
            (int)cR, 2 * cD, cD);
    gemm_bf16_kernel<0, true, true>
        <<<mt * (cD / 64) / 8, 256, 0, stream>>>(
            ffh, 2 * cD, w_f2, 2 * cD, ff2b, xr, cD, xsb, cD,
            (int)cR, cD, 2 * cD);

    // ---- CfC recurrence over T (sequential) ----
    zero_bf16_kernel<<<(cN * cD + 255) / 256, 256, 0, stream>>>(hA, cN * cD);
    for (int t = 0; t < cT; ++t) {
      const bf16_t* hp = (t & 1) ? hB : hA;
      bf16_t* hn = (t & 1) ? hA : hB;
      cell_step_kernel<<<8, 256, 0, stream>>>(
          xsb, hp,
          w_c1 + (size_t)l * cD * 2 * cD, cell_ff1_b + l * cD,
          w_c2 + (size_t)l * cD * 2 * cD, cell_ff2_b + l * cD,
          w_ct + (size_t)l * cD * 2 * cD, cell_tb_b + l * cD,
          cell_logtau + l * cD, dts, t,
          (l == 0) ? nullptr : sin, sout, hn);
    }
  }
}
